// CANNLinear_37898791419997
// MI455X (gfx1250) — matrix operand order repaired and checked by reading, not compiled, not run on hardware
//
#include <hip/hip_runtime.h>

typedef __attribute__((ext_vector_type(16))) _Float16 v16h;
typedef __attribute__((ext_vector_type(8)))  _Float16 v8h;
typedef __attribute__((ext_vector_type(4)))  _Float16 v4h;
typedef __attribute__((ext_vector_type(8)))  float    v8f;

static constexpr int NROWS = 8192;   // batch
static constexpr int KDIM  = 4096;   // in features
static constexpr int NOUT  = 16384;  // out features

static constexpr int BM = 256, BN = 256, BK = 32;
static constexpr int LDT = 40;       // LDS row stride in halves: 32 + 8 pad (80B, 16B-aligned)
static constexpr int GTHREADS = 512; // 16 waves (wave32)
static constexpr int NTILES = KDIM / BK;

// ---------------------------------------------------------------- init header
__global__ void k_init(unsigned* hdr) {
  if (threadIdx.x == 0) hdr[0] = 0u;
}

// ------------------------------------------------- max|W| (uint-bits atomicMax)
__global__ void k_maxabs(const float4* __restrict__ w4, unsigned* __restrict__ hdr,
                         long long n4) {
  long long i = (long long)blockIdx.x * blockDim.x + threadIdx.x;
  const long long stride = (long long)gridDim.x * blockDim.x;
  float m = 0.0f;
  for (; i < n4; i += stride) {
    float4 v = w4[i];
    m = fmaxf(m, fmaxf(fmaxf(fabsf(v.x), fabsf(v.y)), fmaxf(fabsf(v.z), fabsf(v.w))));
  }
  #pragma unroll
  for (int off = 16; off > 0; off >>= 1)
    m = fmaxf(m, __shfl_down(m, off, 32));
  if ((threadIdx.x & 31) == 0)
    atomicMax(hdr, __float_as_uint(m));   // valid: m >= 0
}

// ------------------------------------------------------- quantize W -> int8 q
__global__ void k_quant(const float4* __restrict__ w4, char4* __restrict__ q4,
                        const unsigned* __restrict__ hdr, long long n4) {
  const float scale = fmaxf(__uint_as_float(hdr[0]) * (1.0f / 7.0f), 1e-8f);
  const float inv = 1.0f / scale;
  long long i = (long long)blockIdx.x * blockDim.x + threadIdx.x;
  const long long stride = (long long)gridDim.x * blockDim.x;
  for (; i < n4; i += stride) {
    float4 v = w4[i];
    float a = fminf(fmaxf(rintf(v.x * inv), -8.0f), 7.0f);
    float b = fminf(fmaxf(rintf(v.y * inv), -8.0f), 7.0f);
    float c = fminf(fmaxf(rintf(v.z * inv), -8.0f), 7.0f);
    float d = fminf(fmaxf(rintf(v.w * inv), -8.0f), 7.0f);
    char4 o;
    o.x = (signed char)a; o.y = (signed char)b;
    o.z = (signed char)c; o.w = (signed char)d;
    q4[i] = o;
  }
}

// ----------------------------------------------------------------- x -> f16
__global__ void k_x2h(const float4* __restrict__ x4, v4h* __restrict__ xh4,
                      long long n4) {
  long long i = (long long)blockIdx.x * blockDim.x + threadIdx.x;
  const long long stride = (long long)gridDim.x * blockDim.x;
  for (; i < n4; i += stride) {
    float4 v = x4[i];
    v4h h;
    h[0] = (_Float16)v.x; h[1] = (_Float16)v.y;
    h[2] = (_Float16)v.z; h[3] = (_Float16)v.w;
    xh4[i] = h;
  }
}

// ---- CDNA5 async copy helpers (no probed builtin; ISA §10 / §15.18.3) -------
__device__ __forceinline__ void async_copy_32B(unsigned lds_off,
                                               unsigned long long gaddr) {
  // copies 32 contiguous bytes: two GLOBAL_LOAD_ASYNC_TO_LDS_B128
  // (INST_OFFSET is added to BOTH the LDS and global address per ISA pseudocode)
  asm volatile("global_load_async_to_lds_b128 %0, %1, off\n\t"
               "global_load_async_to_lds_b128 %0, %1, off offset:16"
               :: "v"(lds_off), "v"(gaddr) : "memory");
}
__device__ __forceinline__ void wait_async0() {
  asm volatile("s_wait_asynccnt 0x0" ::: "memory");
}

// --------------------------------------------------------- WMMA f16 GEMM
// out[m][n] = scale * sum_k xh[m][k] * q[n][k] + bias[n]
// Double-buffered LDS, async global->LDS for the f16 A tile, register-path
// int8->f16 expansion for the B tile, one barrier per K-step.
__global__ __launch_bounds__(GTHREADS)
void k_gemm(const _Float16* __restrict__ xh, const signed char* __restrict__ qw,
            const float* __restrict__ bias, const unsigned* __restrict__ hdr,
            float* __restrict__ out) {
  __shared__ _Float16 As[2][BM * LDT];   // 2 x 20 KB
  __shared__ _Float16 Bs[2][BN * LDT];   // 2 x 20 KB

  const int tid  = threadIdx.x;
  const int lane = tid & 31;
  const int wave = tid >> 5;
  const int wm = (wave >> 2) * 64;     // wave M offset in block tile
  const int wn = (wave & 3) * 64;      // wave N offset in block tile
  const int m0 = blockIdx.y * BM;
  const int n0 = blockIdx.x * BN;

  // cooperative tile loaders: thread t -> row t/2, half-row (t&1)*16
  const int lrow = tid >> 1;           // 0..255
  const int lcol = (tid & 1) << 4;     // 0 or 16 (halves / bytes)
  const int ldst = lrow * LDT + lcol;  // LDS element index of this thread's slice

  // A: global byte address of this thread's 32B slice, advanced 64B per tile
  unsigned long long ga =
      (unsigned long long)(uintptr_t)(xh + (long long)(m0 + lrow) * KDIM + lcol);
  const signed char* bptr = qw + (long long)(n0 + lrow) * KDIM + lcol;

  // LDS byte offsets (generic shared address truncates to LDS offset)
  const unsigned ldsA[2] = {
      (unsigned)(uintptr_t)&As[0][ldst],
      (unsigned)(uintptr_t)&As[1][ldst]};

  const v8f vzero = {0.f, 0.f, 0.f, 0.f, 0.f, 0.f, 0.f, 0.f};
  v8f acc[4][4];
  #pragma unroll
  for (int i = 0; i < 4; ++i)
    #pragma unroll
    for (int j = 0; j < 4; ++j)
      acc[i][j] = vzero;

  // WMMA fragment addressing (wave32; contiguous k order chosen here, same for A and B):
  // A 16x32 f16: lane<16 -> K 0..15        ; lane>=16 -> K 16..31
  // B 32x16 f16: lane<16 -> K 0..15        ; lane>=16 -> K 16..31
  const int frow = lane & 15;
  const int ha = (lane < 16) ? 0 : 16;
  const int hb = (lane < 16) ? 0 : 16;

  // ---------------- prologue: tile 0 into buffer 0 ----------------
  async_copy_32B(ldsA[0], ga);
  {
    int4 braw = *(const int4*)bptr;
    v8h b0, b1;
    const signed char* pb = (const signed char*)&braw;
    #pragma unroll
    for (int e = 0; e < 8; ++e) {
      b0[e] = (_Float16)(float)pb[e];
      b1[e] = (_Float16)(float)pb[e + 8];
    }
    *(v8h*)&Bs[0][ldst]     = b0;      // direct shared indexing -> ds_store_b128
    *(v8h*)&Bs[0][ldst + 8] = b1;
  }

  for (int t = 0; t < NTILES; ++t) {
    const int cur = t & 1;
    const int nxt = cur ^ 1;

    wait_async0();       // our async writes to buffer `cur` have landed in LDS
    __syncthreads();     // publish all writes; all reads of buffer `nxt` done

    if (t + 1 < NTILES) {
      // issue next tile while computing current one
      async_copy_32B(ldsA[nxt], ga + (unsigned long long)(t + 1) * (BK * 2));
      int4 braw = *(const int4*)(bptr + (long long)(t + 1) * BK);
      v8h b0, b1;
      const signed char* pb = (const signed char*)&braw;
      #pragma unroll
      for (int e = 0; e < 8; ++e) {                 // int8 -> f16 (exact)
        b0[e] = (_Float16)(float)pb[e];
        b1[e] = (_Float16)(float)pb[e + 8];
      }
      *(v8h*)&Bs[nxt][ldst]     = b0;  // direct shared indexing -> ds_store_b128
      *(v8h*)&Bs[nxt][ldst + 8] = b1;
    }

    // ---------------- compute on buffer `cur` ----------------
    v16h af[4], bf[4];
    #pragma unroll
    for (int i = 0; i < 4; ++i) {
      const _Float16* p = &As[cur][(wm + i * 16 + frow) * LDT + ha];
      v8h lo = *(const v8h*)p;
      v8h hi = *(const v8h*)(p + 8);
      af[i] = __builtin_shufflevector(lo, hi, 0,1,2,3,4,5,6,7,8,9,10,11,12,13,14,15);
    }
    #pragma unroll
    for (int j = 0; j < 4; ++j) {
      const _Float16* p = &Bs[cur][(wn + j * 16 + frow) * LDT + hb];
      v8h lo = *(const v8h*)p;
      v8h hi = *(const v8h*)(p + 8);
      bf[j] = __builtin_shufflevector(lo, hi, 0,1,2,3,4,5,6,7,8,9,10,11,12,13,14,15);
    }

    #pragma unroll
    for (int i = 0; i < 4; ++i)
      #pragma unroll
      for (int j = 0; j < 4; ++j)
        acc[i][j] = __builtin_amdgcn_wmma_f32_16x16x32_f16(
            false, af[i], false, bf[j], (short)0, acc[i][j], false, false);
  }

  // epilogue: D layout (ISA 7.12.2): lane<16 -> M = r, lane>=16 -> M = r+8; N = lane&15
  const float scale = fmaxf(__uint_as_float(hdr[0]) * (1.0f / 7.0f), 1e-8f);
  const int mofs = (lane < 16) ? 0 : 8;
  const int ncol = lane & 15;
  #pragma unroll
  for (int j = 0; j < 4; ++j) {
    const int n = n0 + wn + j * 16 + ncol;
    const float bv = bias[n];
    #pragma unroll
    for (int i = 0; i < 4; ++i) {
      const int mb = m0 + wm + i * 16 + mofs;
      #pragma unroll
      for (int r = 0; r < 8; ++r)
        out[(long long)(mb + r) * NOUT + n] = acc[i][j][r] * scale + bv;
    }
  }
}

// ---------------------------------------------------------------------- launch
extern "C" void kernel_launch(void* const* d_in, const int* in_sizes, int n_in,
                              void* d_out, int out_size, void* d_ws, size_t ws_size,
                              hipStream_t stream) {
  (void)in_sizes; (void)n_in; (void)out_size; (void)ws_size;
  const float* x    = (const float*)d_in[0];
  const float* w    = (const float*)d_in[1];
  const float* bias = (const float*)d_in[2];
  float* out = (float*)d_out;

  // ws: [0..63] header (max|W| bits) | int8 q [NOUT*KDIM] | f16 x [NROWS*KDIM]
  unsigned*    hdr = (unsigned*)d_ws;
  signed char* qw  = (signed char*)d_ws + 64;
  _Float16*    xh  = (_Float16*)((char*)d_ws + 64 + (size_t)NOUT * KDIM);

  k_init  <<<1, 64, 0, stream>>>(hdr);
  k_maxabs<<<2048, 256, 0, stream>>>((const float4*)w, hdr, (long long)NOUT * KDIM / 4);
  k_quant <<<4096, 256, 0, stream>>>((const float4*)w, (char4*)qw, hdr,
                                     (long long)NOUT * KDIM / 4);
  k_x2h   <<<2048, 256, 0, stream>>>((const float4*)x, (v4h*)xh,
                                     (long long)NROWS * KDIM / 4);

  dim3 grid(NOUT / BN, NROWS / BM);   // 64 x 32 blocks
  k_gemm<<<grid, GTHREADS, 0, stream>>>(xh, qw, bias, hdr, out);
}
